// CustomRNN_85839216378390
// MI455X (gfx1250) — matrix operand order repaired and checked by reading, not compiled, not run on hardware
//
#include <hip/hip_runtime.h>
#include <hip/hip_bf16.h>

// CDNA5 / gfx1250: wave32, WMMA bf16 path.
typedef __bf16 v16bf __attribute__((ext_vector_type(16)));
typedef float  v8f   __attribute__((ext_vector_type(8)));

#define RNN_B 64
#define RNN_S 2048
#define RNN_I 512
#define RNN_H 512
#define NUM_SCAN_BLOCKS 32

// ---------------------------------------------------------------------------
// Zero the device barrier counter (graph-capturable, deterministic per launch)
// ---------------------------------------------------------------------------
__global__ void zero_ctr_kernel(unsigned int* ctr) {
    if (threadIdx.x == 0 && blockIdx.x == 0) *ctr = 0u;
}

// ---------------------------------------------------------------------------
// Transpose W (IxH) and U (HxH) into bf16 column-major (WT[h][i], UT[h][k]).
// Makes per-lane B-fragment loads (one column, 16 consecutive K) contiguous.
// ---------------------------------------------------------------------------
__global__ __launch_bounds__(256)
void prep_weights_kernel(const float* __restrict__ W, const float* __restrict__ U,
                         unsigned short* __restrict__ WTraw, unsigned short* __restrict__ UTraw) {
    __bf16* WT = reinterpret_cast<__bf16*>(WTraw);
    __bf16* UT = reinterpret_cast<__bf16*>(UTraw);
    const int n = RNN_I * RNN_H;
    int idx = blockIdx.x * blockDim.x + threadIdx.x;
    if (idx < n) {
        int i = idx / RNN_H, h = idx % RNN_H;
        WT[(size_t)h * RNN_I + i] = (__bf16)W[idx];
    } else if (idx < 2 * n) {
        int k = idx - n;
        int i = k / RNN_H, h = k % RNN_H;
        UT[(size_t)h * RNN_H + i] = (__bf16)U[k];
    }
}

// --- fragment loaders (A: f32 rows -> bf16; B: contiguous bf16 columns) ----
__device__ __forceinline__ void load_a_frag(const float* __restrict__ xrow,
                                            int k, int khalfA, v16bf& a) {
#pragma unroll
    for (int e = 0; e < 8; ++e) a[e]     = (__bf16)xrow[k + khalfA + e];
#pragma unroll
    for (int e = 0; e < 8; ++e) a[8 + e] = (__bf16)xrow[k + khalfA + 8 + e];
}

__device__ __forceinline__ void load_b_frags(const __bf16* __restrict__ bbase,
                                             int k, v16bf (&b)[4]) {
#pragma unroll
    for (int t = 0; t < 4; ++t) {
        const __bf16* p = bbase + (size_t)(t * 16) * RNN_I + k;
#pragma unroll
        for (int e = 0; e < 16; ++e) b[t][e] = p[e];
    }
}

// ---------------------------------------------------------------------------
// Phase 1: xW[b*S+s][h] = x[b,s,:] . W[:,h] + bias[h], stored bf16.
// One wave per 16-row tile x 4 N-tiles (A-fragment reuse = 4).
// k-loop unrolled by 2 with ping-pong fragment buffers: the "next" loads
// write directly into the set consumed one stage later, so there are no
// loop-carried register copies (kills the v_dual_mov churn of the
// single-buffer rotation) while loads stay in flight under the WMMAs.
// ---------------------------------------------------------------------------
__global__ __launch_bounds__(256)
void xw_gemm_kernel(const float* __restrict__ x, const unsigned short* __restrict__ WTraw,
                    const float* __restrict__ bias, unsigned short* __restrict__ xWraw) {
    const __bf16* WT = reinterpret_cast<const __bf16*>(WTraw);
    __bf16* xW = reinterpret_cast<__bf16*>(xWraw);

    const int lane  = threadIdx.x & 31;
    const int wave  = threadIdx.x >> 5;                  // 0..7
    const int gwave = blockIdx.x * 8 + wave;
    const int row_tile = gwave >> 3;                     // 0..8191  (M/16)
    const int colgrp   = gwave & 7;                      // 0..7 -> 4 N-tiles each
    const int m0 = row_tile * 16;
    const int n0 = colgrp * 64;

    const int l15    = lane & 15;
    const int khalfA = (lane < 16) ? 0 : 16;             // A k order: contiguous k = 16*(lane/16) + e
    const int kbaseB = (lane < 16) ? 0 : 16;             // B k order: contiguous, same as A

    v8f acc[4] = {};                                     // f32 accumulators
    const float* xrow = x + (size_t)(m0 + l15) * RNN_I;  // A row m = lane&15
    const __bf16* bbase = WT + (size_t)(n0 + l15) * RNN_I + kbaseB;

    // --- prologue: fragments for k = 0 into buffer 0 ---
    v16bf a0, b0[4], a1, b1[4];
    load_a_frag(xrow, 0, khalfA, a0);
    load_b_frags(bbase, 0, b0);

    for (int k = 0; k < RNN_I; k += 64) {
        // stage 1: load k+32 into buffer 1 (always in-bounds: k+32 <= 480)
        load_a_frag(xrow, k + 32, khalfA, a1);
        load_b_frags(bbase, k + 32, b1);
        __builtin_prefetch(xrow + k + 128, 0, 1);        // global_prefetch_b8
#pragma unroll
        for (int t = 0; t < 4; ++t)
            acc[t] = __builtin_amdgcn_wmma_f32_16x16x32_bf16(
                false, a0, false, b0[t], (short)0, acc[t], false, false);

        // stage 2: load k+64 into buffer 0 (clamped on final iteration)
        const int kn = (k + 64 < RNN_I) ? (k + 64) : 0;
        load_a_frag(xrow, kn, khalfA, a0);
        load_b_frags(bbase, kn, b0);
#pragma unroll
        for (int t = 0; t < 4; ++t)
            acc[t] = __builtin_amdgcn_wmma_f32_16x16x32_bf16(
                false, a1, false, b1[t], (short)0, acc[t], false, false);
    }

    // Add bias, store bf16 tiles. C/D layout: lane n=lane&15, vgpr r -> m = r + 8*(lane>=16)
#pragma unroll
    for (int t = 0; t < 4; ++t) {
        const int ncol = n0 + t * 16 + l15;
        const float bv = bias[ncol];
#pragma unroll
        for (int r = 0; r < 8; ++r) {
            const int m = m0 + r + ((lane < 16) ? 0 : 8);
            xW[(size_t)m * RNN_H + ncol] = (__bf16)(acc[t][r] + bv);
        }
    }
}

// ---------------------------------------------------------------------------
// Phase 2: persistent scan. 32 blocks x 128 threads = 128 waves, one 16x16
// output tile per wave per step, device-scope barrier between steps.
// KEY: each wave's U-column fragments for the whole K=512 reduction are only
// 16 x v16bf = 128 VGPRs -> loaded ONCE into registers before the time loop
// (occupancy is irrelevant for a 32-block persistent kernel). Two
// accumulator chains halve the WMMA->WMMA RAW dependency depth.
// ---------------------------------------------------------------------------
__global__ __launch_bounds__(128)
void rnn_scan_kernel(const unsigned short* __restrict__ xWraw,
                     const unsigned short* __restrict__ UTraw,
                     float* __restrict__ out,
                     unsigned int* __restrict__ ctr) {
    const __bf16* xW = reinterpret_cast<const __bf16*>(xWraw);
    const __bf16* UT = reinterpret_cast<const __bf16*>(UTraw);

    __shared__ float tile[4][16][17];                    // per-wave 16x16 (+pad)

    const int lane  = threadIdx.x & 31;
    const int wave  = threadIdx.x >> 5;                  // 0..3
    const int gtile = blockIdx.x * 4 + wave;             // 0..127
    const int mb = gtile >> 5;                           // batch block 0..3
    const int nb = gtile & 31;                           // H block 0..31
    const int m0 = mb * 16, n0 = nb * 16;

    const int l15    = lane & 15;
    const int khalfA = (lane < 16) ? 0 : 16;
    const int kbaseB = (lane < 16) ? 0 : 16;
    const int ncol   = n0 + l15;                         // this lane's output column
    const int brow   = m0 + l15;                         // this lane's A row (batch idx)
    const int mrow   = (lane < 16) ? 0 : 8;              // C/D row base for this lane

    // --- U column fragments, register-resident for the whole scan ---
    v16bf ufrag[16];
    {
        const __bf16* bcol = UT + (size_t)ncol * RNN_H + kbaseB;
#pragma unroll
        for (int kb = 0; kb < 16; ++kb) {
            const __bf16* bptr = bcol + kb * 32;
#pragma unroll
            for (int e = 0; e < 16; ++e) ufrag[kb][e] = bptr[e];
        }
    }

    for (int t = 0; t < RNN_S; ++t) {
        // Pre-issue the xw tile loads so they overlap the WMMA chain.
        float xwv[8];
#pragma unroll
        for (int r = 0; r < 8; ++r) {
            const size_t xwrow = (size_t)(m0 + mrow + r) * RNN_S + (size_t)t;
            xwv[r] = (float)xW[xwrow * RNN_H + ncol];
        }

        v8f acc0 = {}, acc1 = {};
        if (t > 0) {
            // A = h_{t-1}: out[brow][t-1][:], f32 -> bf16 on the fly
            const float* hrow = out + (size_t)brow * (RNN_S * RNN_H) + (size_t)(t - 1) * RNN_H;
#pragma unroll
            for (int kb = 0; kb < 16; ++kb) {
                const int k = kb * 32;
                v16bf a;
#pragma unroll
                for (int e = 0; e < 8; ++e) a[e]     = (__bf16)hrow[k + khalfA + e];
#pragma unroll
                for (int e = 0; e < 8; ++e) a[8 + e] = (__bf16)hrow[k + khalfA + 8 + e];
                if (kb & 1)
                    acc1 = __builtin_amdgcn_wmma_f32_16x16x32_bf16(
                        false, a, false, ufrag[kb], (short)0, acc1, false, false);
                else
                    acc0 = __builtin_amdgcn_wmma_f32_16x16x32_bf16(
                        false, a, false, ufrag[kb], (short)0, acc0, false, false);
            }
        }

        // h = tanh(acc + xw_t); stage in LDS (WMMA layout -> row-major)
#pragma unroll
        for (int r = 0; r < 8; ++r) {
            tile[wave][mrow + r][l15] = tanhf(acc0[r] + acc1[r] + xwv[r]);
        }
        __syncthreads();
        // Coalesced stores: half-wave per row, 16 consecutive floats
#pragma unroll
        for (int rr = 0; rr < 8; ++rr) {
            const int m = rr * 2 + (lane >> 4);
            out[(size_t)(m0 + m) * (RNN_S * RNN_H) + (size_t)t * RNN_H + n0 + l15]
                = tile[wave][m][l15];
        }
        __syncthreads();

        if (t < RNN_S - 1) {
            __threadfence();                              // release h_t to device scope
            __syncthreads();
            if (threadIdx.x == 0) {
                __hip_atomic_fetch_add(ctr, 1u, __ATOMIC_RELEASE, __HIP_MEMORY_SCOPE_AGENT);
                const unsigned target = (unsigned)(t + 1) * NUM_SCAN_BLOCKS;
                while (__hip_atomic_load(ctr, __ATOMIC_ACQUIRE, __HIP_MEMORY_SCOPE_AGENT) < target) {
                    __builtin_amdgcn_s_sleep(1);
                }
            }
            __syncthreads();
        }
    }
}

// ---------------------------------------------------------------------------
// h_last = hidden_seq[:, S-1, :]
// ---------------------------------------------------------------------------
__global__ __launch_bounds__(256)
void copy_hlast_kernel(float* __restrict__ out) {
    int idx = blockIdx.x * blockDim.x + threadIdx.x;     // 0..B*H-1
    if (idx < RNN_B * RNN_H) {
        const int b = idx / RNN_H, h = idx % RNN_H;
        out[(size_t)RNN_B * RNN_S * RNN_H + idx] =
            out[(size_t)b * RNN_S * RNN_H + (size_t)(RNN_S - 1) * RNN_H + h];
    }
}

// ---------------------------------------------------------------------------
extern "C" void kernel_launch(void* const* d_in, const int* in_sizes, int n_in,
                              void* d_out, int out_size, void* d_ws, size_t ws_size,
                              hipStream_t stream) {
    const float* x  = (const float*)d_in[0];   // [B,S,I]
    const float* Wi = (const float*)d_in[1];   // [I,H]
    const float* Ui = (const float*)d_in[2];   // [H,H]
    const float* bi = (const float*)d_in[3];   // [H]
    float* out = (float*)d_out;                // [B,S,H] + [B,H]

    // Workspace layout (bf16 stored as u16 at the ABI boundary):
    //   [0, 512KB)           : WT  bf16 [H][I]
    //   [512KB, 1MB)         : UT  bf16 [H][H]
    //   [1MB, 1MB+128MB)     : xW  bf16 [B*S][H]
    //   then                 : barrier counter (u32)
    char* ws = (char*)d_ws;
    unsigned short* WT = (unsigned short*)ws;
    unsigned short* UT = (unsigned short*)(ws + (size_t)RNN_I * RNN_H * 2);
    unsigned short* xW = (unsigned short*)(ws + (size_t)2 * RNN_I * RNN_H * 2);
    unsigned int*  ctr = (unsigned int*)(ws + (size_t)2 * RNN_I * RNN_H * 2
                                            + (size_t)RNN_B * RNN_S * RNN_H * 2);

    zero_ctr_kernel<<<1, 32, 0, stream>>>(ctr);

    {   // weight transpose + bf16 convert
        const int total = 2 * RNN_I * RNN_H;
        prep_weights_kernel<<<(total + 255) / 256, 256, 0, stream>>>(Wi, Ui, WT, UT);
    }
    {   // xW = x@W + b  (131072x512x512, bf16 WMMA, 4 N-tiles per wave)
        const int row_tiles = (RNN_B * RNN_S) / 16;      // 8192
        const int colgrps   = RNN_H / 64;                // 8
        const int waves     = row_tiles * colgrps;       // 65536
        xw_gemm_kernel<<<waves / 8, 256, 0, stream>>>(x, WT, bi, xW);
    }
    // persistent scan over S with device-scope barrier
    rnn_scan_kernel<<<NUM_SCAN_BLOCKS, 128, 0, stream>>>(xW, UT, out, ctr);

    copy_hlast_kernel<<<(RNN_B * RNN_H + 255) / 256, 256, 0, stream>>>(out);
    (void)in_sizes; (void)n_in; (void)out_size; (void)ws_size;
}
